// GATLayer_18193481466008
// MI455X (gfx1250) — matrix operand order repaired and checked by reading, not compiled, not run on hardware
//
#include <hip/hip_runtime.h>

// GAT layer, N=8192 nodes, F=256 features, MI455X (gfx1250, wave32).
//
// Roofline: mandatory HBM traffic = adj (256 MB). We read it exactly once
// (bit-pack kernel), then all passes work off an 8 MB bitmask + 4 MB bf16
// WhT that stays L2-resident (192 MB L2). Big matmul (attention @ Wh,
// 34.4 GFLOP) runs on v_wmma_f32_16x16x32_bf16 with f32 accumulation;
// Wh = h@W runs on v_wmma_f32_16x16x4_f32 (full fp32).

#define GN 8192          // nodes
#define GF 256           // features
#define NWORD (GN / 32)  // mask words per row
#define NEG_BIG (-9.0e15f)
#define SLOPE 0.2f

typedef __attribute__((ext_vector_type(2)))  float  v2f;
typedef __attribute__((ext_vector_type(8)))  float  v8f;
typedef __attribute__((ext_vector_type(8)))  __bf16 v8bf;
typedef __attribute__((ext_vector_type(16))) __bf16 v16bf;
typedef __attribute__((ext_vector_type(2)))  __bf16 v2bf;

__device__ __forceinline__ float lrelu(float x) {
    // leaky_relu(x, 0.2) == max(x, 0.2*x) for all x
    return fmaxf(x, SLOPE * x);
}

// ---------------------------------------------------------------------------
// Kernel 0: bit-pack adj (fp32 0/1, 256 MB) -> 32-bit masks (8 MB).
// One wave emits 32 mask words per pass; fully coalesced 128B reads/iter.
// ---------------------------------------------------------------------------
__global__ __launch_bounds__(256) void gat_pack(const float* __restrict__ adj,
                                                unsigned* __restrict__ mask) {
    const int t    = blockIdx.x * 256 + threadIdx.x;
    const int lane = t & 31;
    const int wv   = t >> 5;                 // global wave id (65536 total)
    const size_t base = (size_t)wv * 1024;   // 32 words * 32 elements
#pragma unroll 4
    for (int j = 0; j < 32; ++j) {
        float v = adj[base + (size_t)j * 32 + lane];
        unsigned long long b = __ballot(v > 0.0f);   // wave32: low 32 bits
        if (lane == j) mask[wv * 32 + j] = (unsigned)b;
    }
}

// ---------------------------------------------------------------------------
// Kernel 1: Wh = h @ W via V_WMMA_F32_16X16X4_F32 (fp32).
// Block = 256 thr (8 waves) -> 16 rows x 256 cols. Wave w: cols [32w,32w+32).
// Also emits WhT in bf16 (transposed) for contiguous B-fragment loads later.
// ---------------------------------------------------------------------------
__global__ __launch_bounds__(256) void gat_gemm1(const float* __restrict__ h,
                                                 const float* __restrict__ W,
                                                 float* __restrict__ wh,
                                                 __bf16* __restrict__ whT) {
    const int t = threadIdx.x, lane = t & 31, wave = t >> 5;
    const int rowbase = blockIdx.x * 16;
    const int m     = lane & 15;
    const int khalf = (lane >> 4) * 2;   // A/B K sub-offset per lane half
    const int n0    = wave * 32;

    v8f acc[2] = {};
    for (int k0 = 0; k0 < GF; k0 += 4) {
        // A frag: lane m holds h[row=m][k0+khalf .. +1] (contiguous float2)
        v2f a = *(const v2f*)(h + (size_t)(rowbase + m) * GF + k0 + khalf);
#pragma unroll
        for (int ct = 0; ct < 2; ++ct) {
            const int n = n0 + ct * 16 + m;
            v2f b;
            b[0] = W[(size_t)(k0 + khalf + 0) * GF + n];
            b[1] = W[(size_t)(k0 + khalf + 1) * GF + n];
            acc[ct] = __builtin_amdgcn_wmma_f32_16x16x4_f32(
                false, a, false, b, (short)0, acc[ct], false, false);
        }
    }
    // C layout: VGPR v -> row v (+8 for upper lane half), col = lane&15
#pragma unroll
    for (int ct = 0; ct < 2; ++ct) {
#pragma unroll
        for (int v = 0; v < 8; ++v) {
            const int row = rowbase + v + 8 * (lane >> 4);
            const int col = n0 + ct * 16 + m;
            const float x = acc[ct][v];
            wh[(size_t)row * GF + col]  = x;
            whT[(size_t)col * GN + row] = (__bf16)x;
        }
    }
}

// ---------------------------------------------------------------------------
// Kernel 2: s1 = Wh @ a1, s2 = Wh @ a2. One wave per row.
// ---------------------------------------------------------------------------
__global__ __launch_bounds__(256) void gat_svec(const float* __restrict__ wh,
                                                const float* __restrict__ a,
                                                float* __restrict__ s1,
                                                float* __restrict__ s2) {
    const int t = threadIdx.x, lane = t & 31, wave = t >> 5;
    const int row = blockIdx.x * 8 + wave;
    float a1 = 0.f, a2 = 0.f;
#pragma unroll
    for (int c = lane; c < GF; c += 32) {
        const float v = wh[(size_t)row * GF + c];
        a1 += v * a[c];
        a2 += v * a[GF + c];
    }
#pragma unroll
    for (int off = 16; off > 0; off >>= 1) {
        a1 += __shfl_xor(a1, off);
        a2 += __shfl_xor(a2, off);
    }
    if (lane == 0) { s1[row] = a1; s2[row] = a2; }
}

// ---------------------------------------------------------------------------
// Kernel 3: per-row softmax stats (max m_i, denom l_i) over masked
// e_ij = leakyrelu(s1_i + s2_j) (NEG_BIG where adj==0).
// One wave per row; mask words cached in regs; s2 staged in LDS (32 KB).
// Empty rows fall out naturally: m = NEG_BIG -> exp(0)=1 per entry -> uniform.
// ---------------------------------------------------------------------------
__global__ __launch_bounds__(256) void gat_stats(const unsigned* __restrict__ mask,
                                                 const float* __restrict__ s1,
                                                 const float* __restrict__ s2,
                                                 float* __restrict__ mrow,
                                                 float* __restrict__ lrow) {
    __shared__ float s2s[GN];   // 32 KB
    const int t = threadIdx.x;
    for (int i = t; i < GN; i += 256) s2s[i] = s2[i];
    __syncthreads();

    const int lane = t & 31, wave = t >> 5;
    const int row  = blockIdx.x * 8 + wave;
    const float s1v = s1[row];

    unsigned mw[8];
#pragma unroll
    for (int q = 0; q < 8; ++q) mw[q] = mask[(size_t)row * NWORD + lane + 32 * q];

    float mx = NEG_BIG;
#pragma unroll
    for (int q = 0; q < 8; ++q) {
        const int jb = (lane + 32 * q) * 32;
#pragma unroll
        for (int b = 0; b < 32; ++b) {
            const float x = lrelu(s1v + s2s[jb + b]);
            if ((mw[q] >> b) & 1u) mx = fmaxf(mx, x);
        }
    }
#pragma unroll
    for (int off = 16; off > 0; off >>= 1) mx = fmaxf(mx, __shfl_xor(mx, off));

    float sum = 0.f;
#pragma unroll
    for (int q = 0; q < 8; ++q) {
        const int jb = (lane + 32 * q) * 32;
#pragma unroll
        for (int b = 0; b < 32; ++b) {
            const float x = lrelu(s1v + s2s[jb + b]);
            const float e = ((mw[q] >> b) & 1u) ? x : NEG_BIG;
            sum += __expf(e - mx);   // masked -> exp(-huge)=0; empty row -> exp(0)=1
        }
    }
#pragma unroll
    for (int off = 16; off > 0; off >>= 1) sum += __shfl_xor(sum, off);

    if (lane == 0) { mrow[row] = mx; lrow[row] = sum; }
}

// ---------------------------------------------------------------------------
// Kernel 4: fused h_prime = softmax(e) @ Wh, then ELU.
// Block = 256 thr (8 waves) covers 64 rows x 256 cols (4 row-tiles, amortizes
// L2-resident WhT reads). K loop over 32 nodes/iter:
//   - all 256 threads build the 64x32 bf16 P tile in LDS (A-fragment layout),
//     double-buffered, ONE barrier per iter;
//   - each wave: 4 A frags (ds_load_b128 x2 each) + 2 contiguous 32B B frags
//     from WhT, then 8x v_wmma_f32_16x16x32_bf16.
// Epilogue: scale by 1/l_i, ELU, store fp32.
// ---------------------------------------------------------------------------
__global__ __launch_bounds__(256) void gat_attn(const unsigned* __restrict__ mask,
                                                const __bf16* __restrict__ whT,
                                                const float* __restrict__ s1,
                                                const float* __restrict__ s2,
                                                const float* __restrict__ mrow,
                                                const float* __restrict__ lrow,
                                                float* __restrict__ out) {
    __shared__ __bf16 Pt[2][64 * 32];   // 8 KB, double buffered
    const int t = threadIdx.x, lane = t & 31, wave = t >> 5;
    const int rowbase = blockIdx.x * 64;

    // P-compute assignment: thread t -> rows {pr, pr+16, pr+32, pr+48}, k pair pc
    const int pr = t >> 4;
    const int pc = (t & 15) * 2;
    float s1v[4], mv[4];
#pragma unroll
    for (int q = 0; q < 4; ++q) {
        s1v[q] = s1[rowbase + pr + 16 * q];
        mv[q]  = mrow[rowbase + pr + 16 * q];
    }

    // WMMA fragment lane roles
    const int arow = lane & 15;          // A: row within 16-row tile
    const int ab   = (lane >> 4) * 16;   // A: K sub-offset
    const int bkb  = (lane >> 4) * 16;   // B: K sub-offset
    const int ncol = wave * 32 + (lane & 15);

    v8f acc[4][2] = {};

    for (int it = 0; it < GN / 32; ++it) {
        const int j0  = it * 32;
        const int buf = it & 1;

        // ---- build P tile (8 exps per thread over 4 rows) ----
        const v2f s2p = *(const v2f*)(s2 + j0 + pc);
#pragma unroll
        for (int q = 0; q < 4; ++q) {
            const int r = pr + 16 * q;
            const unsigned mw = mask[(size_t)(rowbase + r) * NWORD + it];
            float x0 = lrelu(s1v[q] + s2p[0]);
            float x1 = lrelu(s1v[q] + s2p[1]);
            const float e0 = ((mw >> pc) & 1u)       ? x0 : NEG_BIG;
            const float e1 = ((mw >> (pc + 1)) & 1u) ? x1 : NEG_BIG;
            v2bf pk;
            pk[0] = (__bf16)__expf(e0 - mv[q]);
            pk[1] = (__bf16)__expf(e1 - mv[q]);
            *(v2bf*)&Pt[buf][r * 32 + pc] = pk;
        }
        __syncthreads();   // single barrier per iter (double buffering)

        // ---- A fragments from LDS (contiguous k = 16*half + s, matches B) ----
        v16bf A[4];
#pragma unroll
        for (int q = 0; q < 4; ++q) {
            v8bf lo = *(v8bf*)&Pt[buf][(arow + 16 * q) * 32 + ab];
            v8bf hi = *(v8bf*)&Pt[buf][(arow + 16 * q) * 32 + 8 + ab];
            A[q] = __builtin_shufflevector(lo, hi, 0, 1, 2, 3, 4, 5, 6, 7,
                                           8, 9, 10, 11, 12, 13, 14, 15);
        }
        // ---- B fragments: contiguous 32B from transposed bf16 Wh ----
        v16bf B[2];
#pragma unroll
        for (int ct = 0; ct < 2; ++ct)
            B[ct] = *(const v16bf*)(whT + (size_t)(ncol + ct * 16) * GN + j0 + bkb);

#pragma unroll
        for (int q = 0; q < 4; ++q)
#pragma unroll
            for (int ct = 0; ct < 2; ++ct)
                acc[q][ct] = __builtin_amdgcn_wmma_f32_16x16x32_bf16(
                    false, A[q], false, B[ct], (short)0, acc[q][ct], false, false);
    }

    // ---- epilogue: normalize by l_i, ELU, store ----
    const int rhalf = 8 * (lane >> 4);
#pragma unroll
    for (int q = 0; q < 4; ++q) {
#pragma unroll
        for (int v = 0; v < 8; ++v) {
            const int row = rowbase + q * 16 + v + rhalf;
            const float linv = 1.0f / lrow[row];
#pragma unroll
            for (int ct = 0; ct < 2; ++ct) {
                float x = acc[q][ct][v] * linv;
                x = x > 0.f ? x : (__expf(x) - 1.0f);   // ELU(alpha=1)
                out[(size_t)row * GF + wave * 32 + ct * 16 + (lane & 15)] = x;
            }
        }
    }
}

// ---------------------------------------------------------------------------
// Launch. Inputs: h(8192x256 f32), adj(8192x8192 f32), W(256x256 f32), a(512 f32).
// Workspace layout (needs ~20.2 MB):
//   [0,8M)    : mask (u32, 8192x256 words)
//   [8M,16M)  : Wh fp32
//   [16M,20M) : WhT bf16 (256 x 8192)
//   [20M,...) : s1, s2, mrow, lrow (4 x 32 KB)
// ---------------------------------------------------------------------------
extern "C" void kernel_launch(void* const* d_in, const int* in_sizes, int n_in,
                              void* d_out, int out_size, void* d_ws, size_t ws_size,
                              hipStream_t stream) {
    const float* h   = (const float*)d_in[0];
    const float* adj = (const float*)d_in[1];
    const float* W   = (const float*)d_in[2];
    const float* a   = (const float*)d_in[3];
    float* out = (float*)d_out;

    char* ws = (char*)d_ws;
    unsigned* mask = (unsigned*)ws;
    float*    wh   = (float*)(ws + (8u << 20));
    __bf16*   whT  = (__bf16*)(ws + (16u << 20));
    float*    s1   = (float*)(ws + (20u << 20));
    float*    s2   = s1 + GN;
    float*    mrow = s2 + GN;
    float*    lrow = mrow + GN;

    gat_pack <<<GN * GN / (256 * 32), 256, 0, stream>>>(adj, mask);
    gat_gemm1<<<GN / 16,              256, 0, stream>>>(h, W, wh, whT);
    gat_svec <<<GN / 8,               256, 0, stream>>>(wh, a, s1, s2);
    gat_stats<<<GN / 8,               256, 0, stream>>>(mask, s1, s2, mrow, lrow);
    gat_attn <<<GN / 64,              256, 0, stream>>>(mask, whT, s1, s2, mrow, lrow, out);
}
